// NQGDecoder_80564996538598
// MI455X (gfx1250) — matrix operand order repaired and checked by reading, not compiled, not run on hardware
//
#include <hip/hip_runtime.h>

// ---------------- problem constants ----------------
#define B_   128
#define L_   400
#define T_   30
#define E_   300
#define He_  512
#define Hd_  512
#define A_   512
#define V_   20000
#define VP_  20032         // V padded to mult of 64 (GEMM N, zero-padded)
#define PAD_ID 0
#define UNK_ID 1
#define SOS_ID 2
#define G3_  1536          // 3*Hd
#define XK_  832           // E+He (812) padded to mult of 32
#define RK_  1344          // E+He+Hd (1324) padded to mult of 32
#define M2_  256           // Hd/2 (maxout output, K for readout GEMM)

typedef __bf16 bf16;
typedef __attribute__((ext_vector_type(16))) __bf16 v16bf;
typedef __attribute__((ext_vector_type(8)))  float  v8f;
typedef int v4i __attribute__((vector_size(16)));

union FragBF { v16bf v; float4 f4[2]; };

#define AS1 __attribute__((address_space(1)))
#define AS3 __attribute__((address_space(3)))

#if __has_builtin(__builtin_amdgcn_global_load_async_to_lds_b128)
#define HAVE_ASYNC_LDS 1
#else
#define HAVE_ASYNC_LDS 0
#endif

// copy 16B global -> LDS (async on gfx1250 if builtin available)
__device__ __forceinline__ void async_cp16(void* lds, const void* g) {
#if HAVE_ASYNC_LDS
  __builtin_amdgcn_global_load_async_to_lds_b128((AS1 v4i*)g, (AS3 v4i*)lds, 0, 0);
#else
  *(float4*)lds = *(const float4*)g;
#endif
}

template <int Ncnt>
__device__ __forceinline__ void wait_async() {
#if HAVE_ASYNC_LDS
#if __has_builtin(__builtin_amdgcn_s_wait_asynccnt)
  __builtin_amdgcn_s_wait_asynccnt(Ncnt);
#else
  asm volatile("s_wait_asynccnt %0" ::"i"(Ncnt));
#endif
#endif
}

// ---------------- WMMA bf16 GEMM: C[M,N] = A[M,K] @ Bt[K,N] (+bias[n]) ----------
// A row-major bf16, Bt row-major bf16 (pre-transposed weight), C f32.
// Requirements: M % 64 == 0, N % 64 == 0, K % 32 == 0.
// grid = (N/64, M/64), block = 128 (4 waves). Block tile 64(M) x 64(N).
// Each wave computes a 16(M) x 64(N) strip: 1 A frag reused across 4 WMMAs.
// B panel (32K x 64N bf16) staged in LDS, double-buffered async global->LDS.
#define TS_ 144                       // LDS row stride (bytes), 16B aligned
#define PANEL_ (32 * TS_)             // one panel: 32 K-rows

__global__ void k_gemm_bf16(const bf16* __restrict__ Am, const bf16* __restrict__ Bt,
                            const float* __restrict__ bias, float* __restrict__ C,
                            int M, int N, int K) {
  __shared__ __align__(16) char panel[2 * PANEL_];
  const int tid   = threadIdx.x;
  const int lane  = tid & 31;
  const int wave  = tid >> 5;
  const int half  = lane >> 4;        // 0: lanes 0-15, 1: lanes 16-31
  const int l16   = lane & 15;
  const int mTile = blockIdx.y * 64 + wave * 16;
  const int nTile = blockIdx.x * 64;

  // stage 32x64 bf16 B panel for K rows [k0, k0+32): 2 x 16B chunks per thread
  auto stage = [&](int k0, char* buf) {
#pragma unroll
    for (int j = 0; j < 2; ++j) {
      int c = tid * 2 + j;            // 0..255
      int row = c >> 3, colc = c & 7; // row = K offset, colc*8 = N offset
      async_cp16(buf + row * TS_ + colc * 16,
                 Bt + (size_t)(k0 + row) * N + nTile + colc * 8);
    }
  };

  v8f acc0 = {}, acc1 = {}, acc2 = {}, acc3 = {};
  // A fragment: lane holds row m = mTile+l16. Dwords hold K pairs:
  //   half0: K 0..7 then 16..23 ; half1: K 8..15 then 24..31 (ISA 16-bit A layout)
  const bf16* arow = Am + (size_t)(mTile + l16) * K + (half ? 8 : 0);

  stage(0, panel);                    // prologue: panel 0 -> buf0
  int cur = 0;
  for (int k0 = 0; k0 < K; k0 += 32) {
    if (k0 + 32 < K) {                // prefetch next panel into other buffer
      stage(k0 + 32, panel + (1 - cur) * PANEL_);
      wait_async<2>();                // current panel done; prefetch in flight
    } else {
      wait_async<0>();
    }
    __syncthreads();                  // panel visible to all 4 waves

    FragBF a;
    const float4* ap = (const float4*)(arow + k0);
    a.f4[0] = ap[0];                  // K = off+0..7
    a.f4[1] = ap[2];                  // K = off+16..23

    const char* brow = panel + cur * PANEL_ + l16 * 2;  // lane = N column; elements over K (same order as A)
    FragBF b0, b1, b2, b3;
#pragma unroll
    for (int s = 0; s < 16; ++s) {
      const int kk = (s < 8) ? (half * 8 + s) : (16 + half * 8 + (s - 8));
      const bf16* bp = (const bf16*)(brow + kk * TS_);
      b0.v[s] = bp[0]; b1.v[s] = bp[16]; b2.v[s] = bp[32]; b3.v[s] = bp[48];
    }
    acc0 = __builtin_amdgcn_wmma_f32_16x16x32_bf16(false, a.v, false, b0.v, (short)0, acc0, false, false);
    acc1 = __builtin_amdgcn_wmma_f32_16x16x32_bf16(false, a.v, false, b1.v, (short)0, acc1, false, false);
    acc2 = __builtin_amdgcn_wmma_f32_16x16x32_bf16(false, a.v, false, b2.v, (short)0, acc2, false, false);
    acc3 = __builtin_amdgcn_wmma_f32_16x16x32_bf16(false, a.v, false, b3.v, (short)0, acc3, false, false);

    __syncthreads();                  // protect buffer before next overwrite
    cur ^= 1;
  }

  // C layout: n = nTile + ns*16 + l16, m = mTile + half*8 + r
  const int nbase = nTile + l16;
#pragma unroll
  for (int ns = 0; ns < 4; ++ns) {
    const int n = nbase + ns * 16;
    const float bv = bias ? bias[n] : 0.0f;
    const v8f acc = ns == 0 ? acc0 : ns == 1 ? acc1 : ns == 2 ? acc2 : acc3;
#pragma unroll
    for (int r = 0; r < 8; ++r) {
      const int m = mTile + half * 8 + r;
      C[(size_t)m * N + n] = acc[r] + bv;
    }
  }
}

// ---------------- conversions / weight prep ----------------
__global__ void k_f32_to_bf16(const float* __restrict__ in, bf16* __restrict__ out, int n) {
  int i = blockIdx.x * 256 + threadIdx.x;
  if (i < n) out[i] = (bf16)in[i];
}

// wt[k*Np + n] = (k < K0 && n < N) ? w[n*K0 + k] : 0  (transpose, pad K0->Kp, N->Np)
__global__ void k_weight_t_pad(const float* __restrict__ w, bf16* __restrict__ wt,
                               int N, int Np, int K0, int Kp) {
  int i = blockIdx.x * 256 + threadIdx.x;
  if (i >= Kp * Np) return;
  int k = i / Np, n = i % Np;
  wt[i] = (k < K0 && n < N) ? (bf16)w[(size_t)n * K0 + k] : (bf16)0.0f;
}

// init: copy initial hidden state (f32 + bf16 mirrors), zero ctx
__global__ void k_init(const float* __restrict__ s_in, float* __restrict__ s,
                       bf16* __restrict__ s_bf, float* __restrict__ ctx) {
  int i = blockIdx.x * 256 + threadIdx.x;
  if (i >= B_ * Hd_) return;
  float v = s_in[i];
  s[i] = v; s_bf[i] = (bf16)v; ctx[i] = 0.0f;
}

// x = [embed[word], ctx, 0-pad] in bf16 ; also stash emb in bf16 for rcat
__global__ void k_build_x(const float* __restrict__ embed, const int* __restrict__ tgt,
                          const float* __restrict__ ctx, bf16* __restrict__ x,
                          bf16* __restrict__ emb_bf, int t) {
  int i = blockIdx.x * 256 + threadIdx.x;
  if (i >= B_ * XK_) return;
  int b = i / XK_, c = i % XK_;
  int w;
  if (t == 0) w = SOS_ID;
  else { int tw = tgt[b * T_ + (t - 1)]; w = (tw >= V_) ? UNK_ID : tw; }
  float v;
  if (c < E_)            { v = embed[(size_t)w * E_ + c]; emb_bf[b * E_ + c] = (bf16)v; }
  else if (c < E_ + He_) { v = ctx[b * He_ + (c - E_)]; }
  else                   { v = 0.0f; }
  x[i] = (bf16)v;
}

// GRU gate fusion: s_new from gi[B,3Hd], gh[B,3Hd], s[B,Hd]; updates s (f32 + bf16)
__global__ void k_gru(const float* __restrict__ gi, const float* __restrict__ gh,
                      float* __restrict__ s, bf16* __restrict__ s_bf) {
  int i = blockIdx.x * 256 + threadIdx.x;
  if (i >= B_ * Hd_) return;
  int b = i / Hd_, h = i % Hd_;
  const float* gib = gi + (size_t)b * G3_;
  const float* ghb = gh + (size_t)b * G3_;
  float ir = gib[h], iz = gib[Hd_ + h], in_ = gib[2 * Hd_ + h];
  float hr = ghb[h], hz = ghb[Hd_ + h], hn  = ghb[2 * Hd_ + h];
  float r = 1.0f / (1.0f + expf(-(ir + hr)));
  float z = 1.0f / (1.0f + expf(-(iz + hz)));
  float n = tanhf(in_ + r * hn);
  float sn = (1.0f - z) * n + z * s[i];
  s[i] = sn; s_bf[i] = (bf16)sn;
}

// e[b,l] = sum_a tanh(encproj[b,l,a] + q[b,a]) * att_v[a], masked -> -inf
// block = 256 (8 waves, one (b,l) per wave), grid = B*L/8
__global__ void k_att_score(const float* __restrict__ encproj, const float* __restrict__ q,
                            const float* __restrict__ attv, const int* __restrict__ src,
                            float* __restrict__ e) {
  int lane = threadIdx.x & 31;
  int wave = threadIdx.x >> 5;
  int idx  = blockIdx.x * 8 + wave;        // b*L + l
  int b    = idx / L_;
  const float* ep = encproj + (size_t)idx * A_;
  const float* qp = q + (size_t)b * A_;
  float sum = 0.0f;
  for (int a = lane; a < A_; a += 32) sum += tanhf(ep[a] + qp[a]) * attv[a];
  for (int off = 16; off; off >>= 1) sum += __shfl_down(sum, off, 32);
  if (lane == 0) e[idx] = (src[idx] == PAD_ID) ? -INFINITY : sum;
}

// per-b: softmax over L, write attn, ctx[b,:] = attn @ enc[b,:,:]   (block=256)
__global__ void k_softmax_ctx(const float* __restrict__ e, const float* __restrict__ enc,
                              float* __restrict__ attn, float* __restrict__ ctx) {
  __shared__ float sattn[L_];
  __shared__ float red[256];
  int b = blockIdx.x, tid = threadIdx.x;
  float mx = -INFINITY;
  for (int l = tid; l < L_; l += 256) mx = fmaxf(mx, e[b * L_ + l]);
  red[tid] = mx; __syncthreads();
  for (int s = 128; s; s >>= 1) { if (tid < s) red[tid] = fmaxf(red[tid], red[tid + s]); __syncthreads(); }
  mx = red[0]; __syncthreads();
  float sum = 0.0f;
  for (int l = tid; l < L_; l += 256) { float v = expf(e[b * L_ + l] - mx); sattn[l] = v; sum += v; }
  red[tid] = sum; __syncthreads();
  for (int s = 128; s; s >>= 1) { if (tid < s) red[tid] += red[tid + s]; __syncthreads(); }
  float inv = 1.0f / red[0]; __syncthreads();
  for (int l = tid; l < L_; l += 256) { float a = sattn[l] * inv; sattn[l] = a; attn[b * L_ + l] = a; }
  __syncthreads();
  for (int c = tid; c < He_; c += 256) {
    const float* ep2 = enc + (size_t)b * L_ * He_ + c;
    float acc = 0.0f;
    for (int l = 0; l < L_; ++l) acc += sattn[l] * ep2[(size_t)l * He_];
    ctx[b * He_ + c] = acc;
  }
}

// p_copy[b] = sigmoid([s,ctx] . copy_W + copy_b)   (one wave per b)
__global__ void k_copy(const float* __restrict__ s, const float* __restrict__ ctx,
                       const float* __restrict__ cw, const float* __restrict__ cb,
                       float* __restrict__ pc) {
  int b = blockIdx.x, lane = threadIdx.x;
  float sum = (lane == 0) ? cb[0] : 0.0f;
  for (int c = lane; c < Hd_; c += 32) sum += s[b * Hd_ + c] * cw[c];
  for (int c = lane; c < He_; c += 32) sum += ctx[b * He_ + c] * cw[Hd_ + c];
  for (int off = 16; off; off >>= 1) sum += __shfl_down(sum, off, 32);
  if (lane == 0) pc[b] = 1.0f / (1.0f + expf(-sum));
}

// rcat = [emb, ctx, s, 0-pad] in bf16
__global__ void k_build_rcat(const bf16* __restrict__ emb_bf, const float* __restrict__ ctx,
                             const float* __restrict__ s, bf16* __restrict__ rcat) {
  int i = blockIdx.x * 256 + threadIdx.x;
  if (i >= B_ * RK_) return;
  int b = i / RK_, c = i % RK_;
  bf16 v;
  if (c < E_)                  v = emb_bf[b * E_ + c];
  else if (c < E_ + He_)       v = (bf16)ctx[b * He_ + (c - E_)];
  else if (c < E_ + He_ + Hd_) v = (bf16)s[b * Hd_ + (c - E_ - He_)];
  else                         v = (bf16)0.0f;
  rcat[i] = v;
}

// maxout(2): m[b,j] = max(rvec[b,2j], rvec[b,2j+1]) -> bf16
__global__ void k_maxout(const float* __restrict__ rvec, bf16* __restrict__ m) {
  int i = blockIdx.x * 256 + threadIdx.x;
  if (i >= B_ * M2_) return;
  int b = i / M2_, j = i % M2_;
  m[i] = (bf16)fmaxf(rvec[b * Hd_ + 2 * j], rvec[b * Hd_ + 2 * j + 1]);
}

// final: softmax over V, mix with copy, write log-probs for step t (block=256, grid=B)
// energy has padded row pitch VP_; only the first V_ columns are read.
__global__ void k_output(const float* __restrict__ energy, const float* __restrict__ attn,
                         const float* __restrict__ pc, float* __restrict__ out, int t) {
  __shared__ float red[256];
  int b = blockIdx.x, tid = threadIdx.x;
  const float* eb = energy + (size_t)b * VP_;
  float mx = -INFINITY;
  for (int v = tid; v < V_; v += 256) mx = fmaxf(mx, eb[v]);
  red[tid] = mx; __syncthreads();
  for (int s = 128; s; s >>= 1) { if (tid < s) red[tid] = fmaxf(red[tid], red[tid + s]); __syncthreads(); }
  mx = red[0]; __syncthreads();
  float sum = 0.0f;
  for (int v = tid; v < V_; v += 256) sum += expf(eb[v] - mx);
  red[tid] = sum; __syncthreads();
  for (int s = 128; s; s >>= 1) { if (tid < s) red[tid] += red[tid + s]; __syncthreads(); }
  sum = red[0]; __syncthreads();
  float p = pc[b];
  float scale = (1.0f - p) / sum;
  float* ob = out + ((size_t)b * T_ + t) * (V_ + L_);
  for (int v = tid; v < V_; v += 256) ob[v] = logf(expf(eb[v] - mx) * scale + 1e-12f);
  for (int l = tid; l < L_; l += 256) ob[V_ + l] = logf(p * attn[b * L_ + l] + 1e-12f);
}

// ---------------- host orchestration ----------------
static inline void gemm(const bf16* A, const bf16* Bt, const float* bias, float* C,
                        int M, int N, int K, hipStream_t s) {
  dim3 g(N / 64, M / 64);
  hipLaunchKernelGGL(k_gemm_bf16, g, dim3(128), 0, s, A, Bt, bias, C, M, N, K);
}

extern "C" void kernel_launch(void* const* d_in, const int* in_sizes, int n_in,
                              void* d_out, int out_size, void* d_ws, size_t ws_size,
                              hipStream_t stream) {
  const float* enc     = (const float*)d_in[0];
  const float* s_in    = (const float*)d_in[1];
  const int*   src     = (const int*)  d_in[2];
  const int*   tgt     = (const int*)  d_in[3];
  const float* embed   = (const float*)d_in[4];
  const float* W_ih    = (const float*)d_in[5];
  const float* b_ih    = (const float*)d_in[6];
  const float* W_hh    = (const float*)d_in[7];
  const float* b_hh    = (const float*)d_in[8];
  const float* att_Wh  = (const float*)d_in[9];
  const float* att_Ws  = (const float*)d_in[10];
  const float* att_b   = (const float*)d_in[11];
  const float* att_v   = (const float*)d_in[12];
  const float* copy_W  = (const float*)d_in[13];
  const float* copy_b  = (const float*)d_in[14];
  const float* read_W  = (const float*)d_in[15];
  const float* read_b  = (const float*)d_in[16];
  const float* read_Wo = (const float*)d_in[17];
  float* out = (float*)d_out;

  // workspace carving (256B aligned)
  char* base = (char*)d_ws;
  size_t off = 0;
  auto alloc = [&](size_t bytes) -> char* {
    char* p = base + off;
    off = (off + bytes + 255) & ~(size_t)255;
    return p;
  };
  bf16*  enc_bf   = (bf16*) alloc((size_t)B_ * L_ * He_ * 2);
  float* encproj  = (float*)alloc((size_t)B_ * L_ * A_ * 4);
  bf16*  wtWh     = (bf16*) alloc((size_t)He_ * A_ * 2);
  bf16*  wtWih    = (bf16*) alloc((size_t)XK_ * G3_ * 2);
  bf16*  wtWhh    = (bf16*) alloc((size_t)Hd_ * G3_ * 2);
  bf16*  wtWs     = (bf16*) alloc((size_t)Hd_ * A_ * 2);
  bf16*  wtRW     = (bf16*) alloc((size_t)RK_ * Hd_ * 2);
  bf16*  wtWo     = (bf16*) alloc((size_t)M2_ * VP_ * 2);
  bf16*  xbuf     = (bf16*) alloc((size_t)B_ * XK_ * 2);
  bf16*  embbf    = (bf16*) alloc((size_t)B_ * E_ * 2);
  bf16*  rcat     = (bf16*) alloc((size_t)B_ * RK_ * 2);
  float* sbuf     = (float*)alloc((size_t)B_ * Hd_ * 4);
  bf16*  sbf      = (bf16*) alloc((size_t)B_ * Hd_ * 2);
  float* ctx      = (float*)alloc((size_t)B_ * He_ * 4);
  float* gi       = (float*)alloc((size_t)B_ * G3_ * 4);
  float* gh       = (float*)alloc((size_t)B_ * G3_ * 4);
  float* qb       = (float*)alloc((size_t)B_ * A_ * 4);
  float* eb       = (float*)alloc((size_t)B_ * L_ * 4);
  float* attn     = (float*)alloc((size_t)B_ * L_ * 4);
  float* pcopy    = (float*)alloc((size_t)B_ * 4);
  float* rvec     = (float*)alloc((size_t)B_ * Hd_ * 4);
  bf16*  mbf      = (bf16*) alloc((size_t)B_ * M2_ * 2);
  float* energy   = (float*)alloc((size_t)B_ * VP_ * 4);
  (void)ws_size; (void)n_in; (void)in_sizes; (void)out_size;

  auto blocks = [](size_t n) { return dim3((unsigned)((n + 255) / 256)); };

  // ---- one-time prep (per call): bf16 conversions + enc_proj GEMM ----
  hipLaunchKernelGGL(k_f32_to_bf16, blocks((size_t)B_ * L_ * He_), dim3(256), 0, stream,
                     enc, enc_bf, B_ * L_ * He_);
  hipLaunchKernelGGL(k_weight_t_pad, blocks((size_t)He_ * A_),   dim3(256), 0, stream, att_Wh, wtWh, A_, A_, He_, He_);
  hipLaunchKernelGGL(k_weight_t_pad, blocks((size_t)XK_ * G3_),  dim3(256), 0, stream, W_ih, wtWih, G3_, G3_, E_ + He_, XK_);
  hipLaunchKernelGGL(k_weight_t_pad, blocks((size_t)Hd_ * G3_),  dim3(256), 0, stream, W_hh, wtWhh, G3_, G3_, Hd_, Hd_);
  hipLaunchKernelGGL(k_weight_t_pad, blocks((size_t)Hd_ * A_),   dim3(256), 0, stream, att_Ws, wtWs, A_, A_, Hd_, Hd_);
  hipLaunchKernelGGL(k_weight_t_pad, blocks((size_t)RK_ * Hd_),  dim3(256), 0, stream, read_W, wtRW, Hd_, Hd_, E_ + He_ + Hd_, RK_);
  hipLaunchKernelGGL(k_weight_t_pad, blocks((size_t)M2_ * VP_),  dim3(256), 0, stream, read_Wo, wtWo, V_, VP_, M2_, M2_);
  hipLaunchKernelGGL(k_init, blocks((size_t)B_ * Hd_), dim3(256), 0, stream, s_in, sbuf, sbf, ctx);
  gemm(enc_bf, wtWh, nullptr, encproj, B_ * L_, A_, He_, stream);

  // ---- sequential decode ----
  for (int t = 0; t < T_; ++t) {
    hipLaunchKernelGGL(k_build_x, blocks((size_t)B_ * XK_), dim3(256), 0, stream,
                       embed, tgt, ctx, xbuf, embbf, t);
    gemm(xbuf, wtWih, b_ih, gi, B_, G3_, XK_, stream);
    gemm(sbf,  wtWhh, b_hh, gh, B_, G3_, Hd_, stream);
    hipLaunchKernelGGL(k_gru, blocks((size_t)B_ * Hd_), dim3(256), 0, stream, gi, gh, sbuf, sbf);
    gemm(sbf, wtWs, att_b, qb, B_, A_, Hd_, stream);
    hipLaunchKernelGGL(k_att_score, dim3(B_ * L_ / 8), dim3(256), 0, stream,
                       encproj, qb, att_v, src, eb);
    hipLaunchKernelGGL(k_softmax_ctx, dim3(B_), dim3(256), 0, stream, eb, enc, attn, ctx);
    hipLaunchKernelGGL(k_copy, dim3(B_), dim3(32), 0, stream, sbuf, ctx, copy_W, copy_b, pcopy);
    hipLaunchKernelGGL(k_build_rcat, blocks((size_t)B_ * RK_), dim3(256), 0, stream,
                       embbf, ctx, sbuf, rcat);
    gemm(rcat, wtRW, read_b, rvec, B_, Hd_, RK_, stream);
    hipLaunchKernelGGL(k_maxout, blocks((size_t)B_ * M2_), dim3(256), 0, stream, rvec, mbf);
    gemm(mbf, wtWo, nullptr, energy, B_, VP_, M2_, stream);
    hipLaunchKernelGGL(k_output, dim3(B_), dim3(256), 0, stream, energy, attn, pcopy, out, t);
  }
}
